// MultiQueryAttention_46385646797344
// MI455X (gfx1250) — matrix operand order repaired and checked by reading, not compiled, not run on hardware
//
#include <hip/hip_runtime.h>
#include <hip/hip_bf16.h>

#define WAVE 32
typedef unsigned short u16;
typedef unsigned int   u32;
typedef __attribute__((ext_vector_type(16))) __bf16 v16bf;
typedef __attribute__((ext_vector_type(8)))  float  v8f;

static constexpr int Bsz  = 4;
static constexpr int Tseq = 2048;
static constexpr int NH   = 16;
static constexpr int HD   = 64;
static constexpr int DM   = 1024;
static constexpr int Mrows = Bsz * Tseq; // 8192

// ---------------- helpers ----------------

__device__ __forceinline__ u16 f2bf(float f) {
  u32 u = __float_as_uint(f);
  u32 r = u + 0x7FFFu + ((u >> 16) & 1u); // round-to-nearest-even
  return (u16)(r >> 16);
}

// A-fragment (16x32 bf16), source row-major with leading dim lda.
// Layout per ISA 7.12.2: lanes 0-15 hold row M=lane, K {0..7,16..23};
// lanes 16-31 hold row M=lane-16, K {8..15,24..31}.
__device__ __forceinline__ v16bf load_a16(const u16* A, int lda, int lane) {
  int m  = lane & 15;
  int kb = (lane & 16) ? 8 : 0;
  const u16* p = A + (size_t)m * lda + kb;
  union { v16bf v; uint4 q[2]; } f;
  f.q[0] = *(const uint4*)(p);
  f.q[1] = *(const uint4*)(p + 16);
  return f.v;
}

// B-fragment (32x16 bf16). Source is "column-major" w.r.t. the B matrix:
// element B[k][n] read from Bm[n*ldb + k] (i.e. Bm row n = output column n,
// contiguous in K). Lanes 0-15: col=lane, K {0..7,16..23}; lanes 16-31:
// col=lane-16, K {8..15,24..31}.
__device__ __forceinline__ v16bf load_b16(const u16* Bm, int ldb, int lane) {
  int n  = lane & 15;
  int kb = (lane & 16) ? 8 : 0;
  const u16* p = Bm + (size_t)n * ldb + kb;
  union { v16bf v; uint4 q[2]; } f;
  f.q[0] = *(const uint4*)(p);
  f.q[1] = *(const uint4*)(p + 16);
  return f.v;
}

__device__ __forceinline__ v8f wmma_bf16(v16bf a, v16bf b, v8f c) {
  return __builtin_amdgcn_wmma_f32_16x16x32_bf16(false, a, false, b,
                                                 (short)0, c, false, false);
}

// C = A(16xK) * W^T for a 16(M) x 64(N) tile. A at (row0,0), W at (col0,0),
// both row-major bf16 with leading dim K.
__device__ __forceinline__ void gemm16x64(const u16* A, const u16* W, int K,
                                          int lane, v8f acc[4]) {
  for (int k = 0; k < K; k += 32) {
    v16bf a = load_a16(A + k, K, lane);
    if (k + 32 < K) {
      __builtin_prefetch(A + k + 32, 0, 1);
      __builtin_prefetch(W + k + 32, 0, 1);
    }
#pragma unroll
    for (int dt = 0; dt < 4; ++dt) {
      v16bf b = load_b16(W + (size_t)(dt * 16) * K + k, K, lane);
      acc[dt] = wmma_bf16(a, b, acc[dt]);
    }
  }
}

// ---------------- kernels ----------------

__global__ void k_cvt(const float* __restrict__ s, u16* __restrict__ d, int n4) {
  int i = blockIdx.x * blockDim.x + threadIdx.x;
  if (i < n4) {
    float4 f = ((const float4*)s)[i];
    uint2 o;
    o.x = (u32)f2bf(f.x) | ((u32)f2bf(f.y) << 16);
    o.y = (u32)f2bf(f.z) | ((u32)f2bf(f.w) << 16);
    ((uint2*)d)[i] = o;
  }
}

// Q projection + interleaved RoPE + 1/sqrt(HD) scaling -> (B,H,T,64) bf16
__global__ __launch_bounds__(WAVE)
void k_gemm_q_rope(const u16* __restrict__ xb, const u16* __restrict__ wq,
                   const float* __restrict__ ct, const float* __restrict__ st,
                   u16* __restrict__ qout) {
  int lane = threadIdx.x;
  int row0 = blockIdx.x * 16;
  int h    = blockIdx.y;
  v8f acc[4] = {};
  gemm16x64(xb + (size_t)row0 * DM, wq + (size_t)h * HD * DM, DM, lane, acc);
  int b  = row0 / Tseq;
  int t0 = row0 % Tseq;
  int rj = (lane & 16) ? 8 : 0;
  int nl = lane & 15;
#pragma unroll
  for (int dt = 0; dt < 4; ++dt) {
    int d = dt * 16 + nl;
#pragma unroll
    for (int j = 0; j < 8; ++j) {
      int t = t0 + j + rj;
      float v  = acc[dt][j];
      float pr = __shfl_xor(v, 1, WAVE);          // partner d^1 (same row)
      float c  = ct[t * HD + d];
      float sn = st[t * HD + d];
      float rot = (d & 1) ? pr : -pr;             // rotate_half_interleaved
      float o = (v * c + rot * sn) * 0.125f;      // fold 1/sqrt(64)
      qout[(((size_t)b * NH + h) * Tseq + t) * HD + d] = f2bf(o);
    }
  }
}

// K (with RoPE, row-major (B,T,64)) and V (transposed (B,64,T)) projections.
__global__ __launch_bounds__(WAVE)
void k_gemm_kv_rope(const u16* __restrict__ xb, const u16* __restrict__ wk,
                    const u16* __restrict__ wv, const float* __restrict__ ct,
                    const float* __restrict__ st, u16* __restrict__ kout,
                    u16* __restrict__ vT) {
  int lane = threadIdx.x;
  int row0 = blockIdx.x * 16;
  bool isV = (blockIdx.y != 0);
  const u16* W = isV ? wv : wk;
  v8f acc[4] = {};
  gemm16x64(xb + (size_t)row0 * DM, W, DM, lane, acc);
  int b  = row0 / Tseq;
  int t0 = row0 % Tseq;
  int rj = (lane & 16) ? 8 : 0;
  int nl = lane & 15;
#pragma unroll
  for (int dt = 0; dt < 4; ++dt) {
    int d = dt * 16 + nl;
#pragma unroll
    for (int j = 0; j < 8; ++j) {
      int t = t0 + j + rj;
      float v = acc[dt][j];
      if (!isV) {
        float pr = __shfl_xor(v, 1, WAVE);
        float c  = ct[t * HD + d];
        float sn = st[t * HD + d];
        float rot = (d & 1) ? pr : -pr;
        kout[((size_t)b * Tseq + t) * HD + d] = f2bf(v * c + rot * sn);
      } else {
        vT[((size_t)b * HD + d) * Tseq + t] = f2bf(v);
      }
    }
  }
}

// Flash-style causal attention: one wave per (b, h, 16-row query tile).
// Processes 32 keys per step: two 16x16 score tiles -> one 16x32 P fragment.
__global__ __launch_bounds__(WAVE)
void k_attn(const u16* __restrict__ qb, const u16* __restrict__ kb,
            const u16* __restrict__ vT, u16* __restrict__ aout) {
  __shared__ u16 pshm[16 * 32];
  int lane = threadIdx.x;
  int t0 = blockIdx.x * 16;
  int h  = blockIdx.y;
  int b  = blockIdx.z;
  const u16* qbase = qb + (((size_t)b * NH + h) * Tseq) * HD;
  const u16* kbase = kb + (size_t)b * Tseq * HD;
  const u16* vbase = vT + (size_t)b * HD * Tseq;

  v16bf aq0 = load_a16(qbase + (size_t)t0 * HD + 0, HD, lane);
  v16bf aq1 = load_a16(qbase + (size_t)t0 * HD + 32, HD, lane);

  v8f accd[4] = {};
  float m[8], l[8];
#pragma unroll
  for (int j = 0; j < 8; ++j) { m[j] = -INFINITY; l[j] = 0.f; }

  int rj = (lane & 16) ? 8 : 0;
  int nl = lane & 15;

  for (int s0 = 0; s0 <= t0 + 15; s0 += 32) {
    // scores = Q * K^T over head dim (two K=32 steps per tile)
    v8f sa = {}, sb = {};
    sa = wmma_bf16(aq0, load_b16(kbase + (size_t)s0 * HD + 0,  HD, lane), sa);
    sa = wmma_bf16(aq1, load_b16(kbase + (size_t)s0 * HD + 32, HD, lane), sa);
    sb = wmma_bf16(aq0, load_b16(kbase + (size_t)(s0 + 16) * HD + 0,  HD, lane), sb);
    sb = wmma_bf16(aq1, load_b16(kbase + (size_t)(s0 + 16) * HD + 32, HD, lane), sb);

    if (s0 + 31 > t0) { // diagonal block: causal mask
#pragma unroll
      for (int j = 0; j < 8; ++j) {
        int row = t0 + j + rj;
        if (s0 + nl      > row) sa[j] = -INFINITY;
        if (s0 + 16 + nl > row) sb[j] = -INFINITY;
      }
    }

#pragma unroll
    for (int j = 0; j < 8; ++j) {
      float mx = fmaxf(sa[j], sb[j]);
#pragma unroll
      for (int off = 1; off < 16; off <<= 1) mx = fmaxf(mx, __shfl_xor(mx, off, WAVE));
      float mn = fmaxf(m[j], mx);
      float fs = __expf(m[j] - mn);
      float pa = __expf(sa[j] - mn);
      float pb = __expf(sb[j] - mn);
      float rs = pa + pb;
#pragma unroll
      for (int off = 1; off < 16; off <<= 1) rs += __shfl_xor(rs, off, WAVE);
      l[j] = l[j] * fs + rs;
      m[j] = mn;
#pragma unroll
      for (int dt = 0; dt < 4; ++dt) accd[dt][j] *= fs;
      int prow = j + rj;
      pshm[prow * 32 + nl]      = f2bf(pa);
      pshm[prow * 32 + 16 + nl] = f2bf(pb);
    }
    __syncthreads();
    // P (16x32) in A layout from LDS, then PV over 4 d-subtiles
    v16bf ap = load_a16(pshm, 32, lane);
#pragma unroll
    for (int dt = 0; dt < 4; ++dt) {
      v16bf bv = load_b16(vbase + (size_t)(dt * 16) * Tseq + s0, Tseq, lane);
      accd[dt] = wmma_bf16(ap, bv, accd[dt]);
    }
    __syncthreads();
  }

#pragma unroll
  for (int dt = 0; dt < 4; ++dt) {
    int d = dt * 16 + nl;
#pragma unroll
    for (int j = 0; j < 8; ++j) {
      int t = t0 + j + rj;
      float o = accd[dt][j] / l[j];
      aout[((size_t)b * Tseq + t) * DM + h * HD + d] = f2bf(o);
    }
  }
}

// Output projection: (B*T,1024)bf16 x Wo^T -> f32
__global__ __launch_bounds__(WAVE)
void k_gemm_out(const u16* __restrict__ ab, const u16* __restrict__ wo,
                float* __restrict__ out) {
  int lane = threadIdx.x;
  int row0 = blockIdx.x * 16;
  int col0 = blockIdx.y * 64;
  v8f acc[4] = {};
  gemm16x64(ab + (size_t)row0 * DM, wo + (size_t)col0 * DM, DM, lane, acc);
  int rj = (lane & 16) ? 8 : 0;
  int nl = lane & 15;
#pragma unroll
  for (int dt = 0; dt < 4; ++dt) {
    int n = col0 + dt * 16 + nl;
#pragma unroll
    for (int j = 0; j < 8; ++j) {
      int row = row0 + j + rj;
      out[(size_t)row * DM + n] = acc[dt][j];
    }
  }
}

// ---------------- launch ----------------

extern "C" void kernel_launch(void* const* d_in, const int* in_sizes, int n_in,
                              void* d_out, int out_size, void* d_ws, size_t ws_size,
                              hipStream_t stream) {
  (void)in_sizes; (void)n_in; (void)out_size; (void)ws_size;
  const float* x  = (const float*)d_in[0];
  const float* Wq = (const float*)d_in[1];
  const float* Wk = (const float*)d_in[2];
  const float* Wv = (const float*)d_in[3];
  const float* Wo = (const float*)d_in[4];
  const float* ct = (const float*)d_in[5];
  const float* st = (const float*)d_in[6];
  // d_in[7] = mask: causal, handled analytically
  float* out = (float*)d_out;

  char* w = (char*)d_ws;
  auto carve = [&](size_t bytes) -> u16* {
    u16* p = (u16*)w;
    w += (bytes + 255) & ~(size_t)255;
    return p;
  };
  u16* xb  = carve((size_t)Mrows * DM * 2);
  u16* wqb = carve((size_t)DM * DM * 2);
  u16* wkb = carve((size_t)HD * DM * 2);
  u16* wvb = carve((size_t)HD * DM * 2);
  u16* wob = carve((size_t)DM * DM * 2);
  u16* qb  = carve((size_t)Mrows * DM * 2);
  u16* kbf = carve((size_t)Mrows * HD * 2);
  u16* vTb = carve((size_t)Mrows * HD * 2);
  u16* ab  = carve((size_t)Mrows * DM * 2);

  auto cvt = [&](const float* src, u16* dst, size_t n) {
    int n4 = (int)(n / 4);
    k_cvt<<<(n4 + 255) / 256, 256, 0, stream>>>(src, dst, n4);
  };
  cvt(x,  xb,  (size_t)Mrows * DM);
  cvt(Wq, wqb, (size_t)DM * DM);
  cvt(Wk, wkb, (size_t)HD * DM);
  cvt(Wv, wvb, (size_t)HD * DM);
  cvt(Wo, wob, (size_t)DM * DM);

  k_gemm_q_rope<<<dim3(Mrows / 16, NH), WAVE, 0, stream>>>(xb, wqb, ct, st, qb);
  k_gemm_kv_rope<<<dim3(Mrows / 16, 2), WAVE, 0, stream>>>(xb, wkb, wvb, ct, st, kbf, vTb);
  k_attn<<<dim3(Tseq / 16, NH, Bsz), WAVE, 0, stream>>>(qb, kbf, vTb, ab);
  k_gemm_out<<<dim3(Mrows / 16, DM / 64), WAVE, 0, stream>>>(ab, wob, out);
}
